// RNN_79113297592496
// MI455X (gfx1250) — matrix operand order repaired and checked by reading, not compiled, not run on hardware
//
#include <hip/hip_runtime.h>
#include <hip/hip_bf16.h>
#include <math.h>

#define B_ 2048
#define T_ 128
#define E_ 256
#define H_ 1024
#define Z_ 256

typedef __bf16 bf16;
typedef __attribute__((ext_vector_type(16))) __bf16 v16bf;
typedef __attribute__((ext_vector_type(8)))  float  v8f;
typedef int v4i __attribute__((ext_vector_type(4)));

#if defined(__gfx1250__) && __has_builtin(__builtin_amdgcn_global_load_async_to_lds_b128)
#define HAVE_ASYNC 1
#else
#define HAVE_ASYNC 0
#endif

__device__ __forceinline__ unsigned short f2bf(float f) {
    union { float f; unsigned int u; } a; a.f = f;
    unsigned int u = a.u;
    u += 0x7fffu + ((u >> 16) & 1u);      // round-to-nearest-even
    return (unsigned short)(u >> 16);
}

union AB {
    v16bf v;
    uint4 q[2];
};

__device__ __forceinline__ float sigmoidf_(float x) {
    return 1.0f / (1.0f + __expf(-x));
}

// Move one 16-byte chunk per lane: global -> LDS (async on CDNA5).
__device__ __forceinline__ void stage16(const bf16* g, char* l) {
#if HAVE_ASYNC
    __builtin_amdgcn_global_load_async_to_lds_b128(
        (__attribute__((address_space(1))) v4i*)g,
        (__attribute__((address_space(3))) v4i*)l, 0, 0);
#else
    *(uint4*)l = *(const uint4*)g;
#endif
}

template <int N>
__device__ __forceinline__ void wait_async() {
#if HAVE_ASYNC
#if __has_builtin(__builtin_amdgcn_s_wait_asynccnt)
    __builtin_amdgcn_s_wait_asynccnt(N);
#else
    asm volatile("s_wait_asynccnt %0" :: "i"(N) : "memory");
#endif
#endif
}

// ---------------------------------------------------------------------------
// Fused LSTM step:  g = [xbf·Wxᵀ] + h_in·Whᵀ + bias ; gate math ; c,h update.
// Block = 256 threads (8 wave32). Block tile: 128 rows × 32 h-cols × 4 gates.
// wave = (wy row-half 0..1, wx gate 0..3); each wave: 4×2 WMMA 16x16 tiles.
// Per 32-wide k-chunk the block stages A (128x32) and B (128x32) tiles into
// LDS in WMMA-fragment order via per-lane async b128 copies, double-buffered,
// single barrier per chunk (wait -> barrier -> issue next -> compute).
// ---------------------------------------------------------------------------
template <bool HAS_X>
__global__ __launch_bounds__(256)
void lstm_step_kernel(const bf16*  __restrict__ xbf,  // [B, Kx] bf16 (HAS_X)
                      int Kx,
                      const bf16*  __restrict__ Wx,   // [4H, Kx] row-major bf16
                      const bf16*  __restrict__ h_in, // [B, H] bf16
                      const bf16*  __restrict__ Wh,   // [4H, H] row-major bf16
                      const float* __restrict__ bias, // [4H]
                      float*       __restrict__ c,    // [B, H] f32 (in/out)
                      bf16*        __restrict__ h_out)// [B, H] bf16
{
    __shared__ __align__(16) union {
        struct { char A[2][8192]; char B[2][8192]; } st;  // staging (32 KB)
        float g[4][128][33];                              // gate spill (66 KB)
    } lds;

    const int lane = threadIdx.x & 31;
    const int wave = threadIdx.x >> 5;
    const int wx   = wave & 3;           // gate index
    const int wy   = wave >> 2;          // row half
    const int bm   = blockIdx.x * 128;   // batch-row base
    const int bn   = blockIdx.y * 32;    // h-column base
    const int l16  = lane & 15;
    const int hi   = lane >> 4;          // 0 or 1

    const int ncx = HAS_X ? (Kx / 32) : 0;
    const int NC  = ncx + H_ / 32;       // total k-chunks across both GEMMs

    // Staging work split (branch-free, wave-uniform):
    const int kh_w = wave & 1;           // which 16-wide k-half this wave stages
    const int tw   = wave >> 1;          // tile group 0..3 (tiles tw and tw+4)

    // Issue all staging copies for virtual chunk i into buffer buf.
    auto issue = [&](int i, int buf) {
        const bf16* A; int lda; const bf16* W; int ldw; int kb;
        if (HAS_X && i < ncx) { A = xbf;  lda = Kx; W = Wx; ldw = Kx; kb = i * 32; }
        else                  { A = h_in; lda = H_; W = Wh; ldw = H_; kb = (i - ncx) * 32; }
        char* dA = lds.st.A[buf];
        char* dB = lds.st.B[buf];
#pragma unroll
        for (int half = 0; half < 2; ++half) {
            // A: tile ta (16 rows x 32 k), fragment chunk kh_w
            const int ta  = tw + half * 4;
            const int row = bm + ta * 16 + l16;
            stage16(A + (size_t)row * lda + kb + kh_w * 16 + hi * 8,
                    dA + ta * 1024 + kh_w * 512 + lane * 16);
            // B: tile tb = gate*2 + colhalf (16 n-rows x 32 k)
            const int tb   = tw + half * 4;
            const int g8   = tb >> 1, ch = tb & 1;
            const int nrow = g8 * H_ + bn + ch * 16 + l16;
            stage16(W + (size_t)nrow * ldw + kb + kh_w * 16 + hi * 8,
                    dB + tb * 1024 + kh_w * 512 + lane * 16);
        }
    };

    v8f acc[4][2] = {};

    issue(0, 0);
    for (int ck = 0; ck < NC; ++ck) {
        const int cur = ck & 1;
        wait_async<0>();      // own chunk-ck copies have landed in LDS
        __syncthreads();      // all waves' copies visible; prev compute done
        if (ck + 1 < NC) issue(ck + 1, cur ^ 1);

        const char* sA = lds.st.A[cur];
        const char* sB = lds.st.B[cur];
        AB bfrag[2];
#pragma unroll
        for (int ct = 0; ct < 2; ++ct) {
            const char* pb = sB + (wx * 2 + ct) * 1024 + lane * 16;
            bfrag[ct].q[0] = *(const uint4*)(pb);
            bfrag[ct].q[1] = *(const uint4*)(pb + 512);
        }
#pragma unroll
        for (int rt = 0; rt < 4; ++rt) {
            AB af;
            const char* pa = sA + (wy * 4 + rt) * 1024 + lane * 16;
            af.q[0] = *(const uint4*)(pa);
            af.q[1] = *(const uint4*)(pa + 512);
#pragma unroll
            for (int ct = 0; ct < 2; ++ct)
                acc[rt][ct] = __builtin_amdgcn_wmma_f32_16x16x32_bf16(
                    false, af.v, false, bfrag[ct].v, (short)0, acc[rt][ct], false, false);
        }
    }
    __syncthreads();   // staging LDS dead; gbuf aliases it

    // ---- spill accumulators to LDS (per C-fragment VGPR layout) ----
#pragma unroll
    for (int rt = 0; rt < 4; ++rt)
#pragma unroll
        for (int ct = 0; ct < 2; ++ct) {
            const int row = wy * 64 + rt * 16 + hi * 8;
            const int col = ct * 16 + l16;
#pragma unroll
            for (int j = 0; j < 8; ++j)
                lds.g[wx][row + j][col] = acc[rt][ct][j];
        }
    __syncthreads();

    // ---- fused LSTM cell elementwise ----
    for (int e = threadIdx.x; e < 128 * 32; e += 256) {
        const int m = e >> 5, col = e & 31;
        const int grow = bm + m, hcol = bn + col;
        const float iv = lds.g[0][m][col] + bias[hcol];
        const float fv = lds.g[1][m][col] + bias[H_ + hcol];
        const float gv = lds.g[2][m][col] + bias[2 * H_ + hcol];
        const float ov = lds.g[3][m][col] + bias[3 * H_ + hcol];
        const size_t idx = (size_t)grow * H_ + hcol;
        const float cn = sigmoidf_(fv) * c[idx] + sigmoidf_(iv) * tanhf(gv);
        c[idx] = cn;
        ((unsigned short*)h_out)[idx] = f2bf(sigmoidf_(ov) * tanhf(cn));
    }
}

// ---------------------------------------------------------------------------
// z = tanh(h·Wzᵀ + bz) -> d_out[:, t, :].  Block tile 128×64, 8 waves.
// ---------------------------------------------------------------------------
__global__ __launch_bounds__(256)
void zproj_kernel(const bf16* __restrict__ h,   // [B, H]
                  const bf16* __restrict__ Wz,  // [Z, H] row-major bf16
                  const float* __restrict__ bz, // [Z]
                  float* __restrict__ out, int t)
{
    const int lane = threadIdx.x & 31;
    const int wave = threadIdx.x >> 5;
    const int wxc  = wave & 3;           // col quarter (16 cols)
    const int wy   = wave >> 2;          // row half (64 rows)
    const int bm   = blockIdx.x * 128;
    const int bn   = blockIdx.y * 64;
    const int l16  = lane & 15;
    const int hi   = lane >> 4;

    v8f acc[4] = {};

    for (int k0 = 0; k0 < H_; k0 += 32) {
        AB bfrag;
        {
            const int nrow = bn + wxc * 16 + l16;
            const uint4* p = (const uint4*)(Wz + (size_t)nrow * H_ + k0 + hi * 8);
            bfrag.q[0] = p[0];
            bfrag.q[1] = p[2];
        }
#pragma unroll
        for (int rt = 0; rt < 4; ++rt) {
            const int m = bm + wy * 64 + rt * 16 + l16;
            const bf16* ap = h + (size_t)m * H_ + k0 + hi * 8;
            AB af;
            af.q[0] = ((const uint4*)ap)[0];
            af.q[1] = ((const uint4*)(ap + 16))[0];
            acc[rt] = __builtin_amdgcn_wmma_f32_16x16x32_bf16(
                false, af.v, false, bfrag.v, (short)0, acc[rt], false, false);
        }
    }

    const int zc = bn + wxc * 16 + l16;
    const float b = bz[zc];
#pragma unroll
    for (int rt = 0; rt < 4; ++rt)
#pragma unroll
        for (int j = 0; j < 8; ++j) {
            const int row = bm + wy * 64 + rt * 16 + hi * 8 + j;
            out[(size_t)row * (T_ * Z_) + (size_t)t * Z_ + zc] = tanhf(acc[rt][j] + b);
        }
}

// ---------------------------------------------------------------------------
// Setup / per-step helper kernels
// ---------------------------------------------------------------------------
__global__ void xcvt_kernel(const float* __restrict__ x, unsigned short* __restrict__ xbf, int t) {
    const int n4 = B_ * E_ / 4;
    int i = blockIdx.x * 256 + threadIdx.x;
    if (i < n4) {
        const int b = (i * 4) / E_;
        const int e = (i * 4) % E_;
        const float4 v = *(const float4*)(x + (size_t)b * (T_ * E_) + (size_t)t * E_ + e);
        uint2 r;
        r.x = (unsigned)f2bf(v.x) | ((unsigned)f2bf(v.y) << 16);
        r.y = (unsigned)f2bf(v.z) | ((unsigned)f2bf(v.w) << 16);
        *(uint2*)(xbf + (size_t)i * 4) = r;
    }
}

__global__ void cvt_f32_bf16(const float* __restrict__ in, unsigned short* __restrict__ out, int n) {
    int i = blockIdx.x * 256 + threadIdx.x;
    if (i < n) out[i] = f2bf(in[i]);
}

__global__ void dec_weights_kernel(const float* __restrict__ Wih, const float* __restrict__ Whh,
                                   unsigned short* __restrict__ Whh_b, unsigned short* __restrict__ Wsum_b, int n) {
    int i = blockIdx.x * 256 + threadIdx.x;
    if (i < n) {
        Whh_b[i]  = f2bf(Whh[i]);
        Wsum_b[i] = f2bf(Wih[i] + Whh[i]);
    }
}

__global__ void bias_prep_kernel(const float* __restrict__ ebih, const float* __restrict__ ebhh,
                                 const float* __restrict__ dbih, const float* __restrict__ dbhh,
                                 const float* __restrict__ dWih,
                                 float* __restrict__ enc_b, float* __restrict__ dec_b,
                                 float* __restrict__ dec0_b) {
    int n = blockIdx.x * 256 + threadIdx.x;
    if (n < 4 * H_) {
        enc_b[n] = ebih[n] + ebhh[n];
        const float db = dbih[n] + dbhh[n];
        dec_b[n] = db;
        float s = 0.f;
        const float* r = dWih + (size_t)n * H_;
        for (int k = 0; k < H_; k += 4)
            s += r[k] + r[k + 1] + r[k + 2] + r[k + 3];
        dec0_b[n] = db + 0.1f * s;   // fold constant 0.1 input of decoder t=0
    }
}

__global__ void init_state_kernel(unsigned short* __restrict__ h0, float* __restrict__ c, int n) {
    int i = blockIdx.x * 256 + threadIdx.x;
    if (i < n) { h0[i] = 0; c[i] = 0.f; }
}

// ---------------------------------------------------------------------------
extern "C" void kernel_launch(void* const* d_in, const int* in_sizes, int n_in,
                              void* d_out, int out_size, void* d_ws, size_t ws_size,
                              hipStream_t stream) {
    (void)in_sizes; (void)n_in; (void)out_size; (void)ws_size;
    const float* x        = (const float*)d_in[0];
    const float* enc_Wih  = (const float*)d_in[1];
    const float* enc_Whh  = (const float*)d_in[2];
    const float* enc_bih  = (const float*)d_in[3];
    const float* enc_bhh  = (const float*)d_in[4];
    const float* dec_Wih  = (const float*)d_in[5];
    const float* dec_Whh  = (const float*)d_in[6];
    const float* dec_bih  = (const float*)d_in[7];
    const float* dec_bhh  = (const float*)d_in[8];
    const float* Wz       = (const float*)d_in[9];
    const float* bz       = (const float*)d_in[10];
    float* out = (float*)d_out;

    char* ws = (char*)d_ws;
    size_t off = 0;
    auto carve = [&](size_t bytes) -> void* {
        void* p = ws + off;
        off += (bytes + 255) & ~(size_t)255;
        return p;
    };
    unsigned short* enc_Wih_b  = (unsigned short*)carve((size_t)4 * H_ * E_ * 2);
    unsigned short* enc_Whh_b  = (unsigned short*)carve((size_t)4 * H_ * H_ * 2);
    unsigned short* dec_Whh_b  = (unsigned short*)carve((size_t)4 * H_ * H_ * 2);
    unsigned short* dec_Wsum_b = (unsigned short*)carve((size_t)4 * H_ * H_ * 2);
    unsigned short* Wz_b       = (unsigned short*)carve((size_t)Z_ * H_ * 2);
    float* enc_b  = (float*)carve((size_t)4 * H_ * 4);
    float* dec_b  = (float*)carve((size_t)4 * H_ * 4);
    float* dec0_b = (float*)carve((size_t)4 * H_ * 4);
    unsigned short* hA  = (unsigned short*)carve((size_t)B_ * H_ * 2);
    unsigned short* hB  = (unsigned short*)carve((size_t)B_ * H_ * 2);
    unsigned short* xbf = (unsigned short*)carve((size_t)B_ * E_ * 2);
    float* c = (float*)carve((size_t)B_ * H_ * 4);

    const dim3 blk(256);
    // --- weight/bias prep ---
    {
        int n1 = 4 * H_ * E_;
        cvt_f32_bf16<<<dim3((n1 + 255) / 256), blk, 0, stream>>>(enc_Wih, enc_Wih_b, n1);
        int n2 = 4 * H_ * H_;
        cvt_f32_bf16<<<dim3((n2 + 255) / 256), blk, 0, stream>>>(enc_Whh, enc_Whh_b, n2);
        int n3 = Z_ * H_;
        cvt_f32_bf16<<<dim3((n3 + 255) / 256), blk, 0, stream>>>(Wz, Wz_b, n3);
        dec_weights_kernel<<<dim3((n2 + 255) / 256), blk, 0, stream>>>(dec_Wih, dec_Whh, dec_Whh_b, dec_Wsum_b, n2);
        bias_prep_kernel<<<dim3((4 * H_ + 255) / 256), blk, 0, stream>>>(
            enc_bih, enc_bhh, dec_bih, dec_bhh, dec_Wih, enc_b, dec_b, dec0_b);
        int ns = B_ * H_;
        init_state_kernel<<<dim3((ns + 255) / 256), blk, 0, stream>>>(hA, c, ns);
    }

    const dim3 gridL(B_ / 128, H_ / 32);   // 16 x 32
    const dim3 gridX((B_ * E_ / 4 + 255) / 256);
    const bf16* hin = (const bf16*)hA;
    bf16* hout = (bf16*)hB;

    // --- encoder: 128 steps ---
    for (int t = 0; t < T_; ++t) {
        xcvt_kernel<<<gridX, blk, 0, stream>>>(x, xbf, t);
        lstm_step_kernel<true><<<gridL, blk, 0, stream>>>(
            (const bf16*)xbf, E_,
            (const bf16*)enc_Wih_b, hin, (const bf16*)enc_Whh_b,
            enc_b, c, hout);
        const bf16* tmp = hin; hin = (const bf16*)hout; hout = (bf16*)tmp;
    }

    // --- decoder: 128 steps (t=0 folds the 0.1 input into dec0_b) ---
    const dim3 gridZ(B_ / 128, Z_ / 64);   // 16 x 4
    for (int t = 0; t < T_; ++t) {
        const bf16* Bmat = (t == 0) ? (const bf16*)dec_Whh_b : (const bf16*)dec_Wsum_b;
        const float* bptr = (t == 0) ? dec0_b : dec_b;
        lstm_step_kernel<false><<<gridL, blk, 0, stream>>>(
            nullptr, 0, nullptr, hin, Bmat, bptr, c, hout);
        zproj_kernel<<<gridZ, blk, 0, stream>>>((const bf16*)hout, (const bf16*)Wz_b, bz, out, t);
        const bf16* tmp = hin; hin = (const bf16*)hout; hout = (bf16*)tmp;
    }
}
